// MultiHeadAttention_75307956568143
// MI455X (gfx1250) — matrix operand order repaired and checked by reading, not compiled, not run on hardware
//
#include <hip/hip_runtime.h>
#include <hip/hip_bf16.h>
#include <stdint.h>

#define NB    2
#define SEQ   2048
#define DM    1024
#define NH    16
#define DHEAD 64

typedef __attribute__((ext_vector_type(16))) __bf16 v16bf;
typedef __attribute__((ext_vector_type(8)))  float  v8f;

union BF16Frag {
    uint4 u4[2];
    v16bf v;
};

__device__ __forceinline__ unsigned short f2bf(float f) {
    union { float f; uint32_t u; } c;
    c.f = f;
    uint32_t u = c.u;
    u += 0x7fffu + ((u >> 16) & 1u);   // round-to-nearest-even
    return (unsigned short)(u >> 16);
}

__device__ __forceinline__ v8f wmma_bf16(const BF16Frag& a, const BF16Frag& b, v8f c) {
    // D = A(16x32 bf16) * B(32x16 bf16) + C(16x16 f32)
    return __builtin_amdgcn_wmma_f32_16x16x32_bf16(
        /*neg_a=*/false, a.v, /*neg_b=*/false, b.v,
        /*c_mod=*/(short)0, c, /*reuse_a=*/false, /*reuse_b=*/false);
}

// A fragment: two 16B chunks at +0 and +16 elements (ISA 16-bit A 16x32 layout)
__device__ __forceinline__ BF16Frag ldfragA(const unsigned short* p) {
    BF16Frag f;
    f.u4[0] = *reinterpret_cast<const uint4*>(p);
    f.u4[1] = *reinterpret_cast<const uint4*>(p + 16);
    return f;
}
// B fragment: two 16B chunks at +0 and +16 elements (same k order as A)
__device__ __forceinline__ BF16Frag ldfragB(const unsigned short* p) {
    BF16Frag f;
    f.u4[0] = *reinterpret_cast<const uint4*>(p);
    f.u4[1] = *reinterpret_cast<const uint4*>(p + 16);
    return f;
}

// ---------------------------------------------------------------------------
// f32 -> bf16 straight conversion, 8 elements per thread
// ---------------------------------------------------------------------------
__global__ void k_cvt(const float* __restrict__ in, unsigned short* __restrict__ out, int n) {
    int i = (blockIdx.x * blockDim.x + threadIdx.x) * 8;
    if (i + 8 <= n) {
        float4 x0 = *reinterpret_cast<const float4*>(in + i);
        float4 x1 = *reinterpret_cast<const float4*>(in + i + 4);
        uint4 r;
        r.x = (uint32_t)f2bf(x0.x) | ((uint32_t)f2bf(x0.y) << 16);
        r.y = (uint32_t)f2bf(x0.z) | ((uint32_t)f2bf(x0.w) << 16);
        r.z = (uint32_t)f2bf(x1.x) | ((uint32_t)f2bf(x1.y) << 16);
        r.w = (uint32_t)f2bf(x1.z) | ((uint32_t)f2bf(x1.w) << 16);
        *reinterpret_cast<uint4*>(out + i) = r;
    }
}

// ---------------------------------------------------------------------------
// f32 [DM][DM] -> bf16 transposed [DM][DM] via 32x32 LDS tile (coalesced both ways)
// ---------------------------------------------------------------------------
__global__ void __launch_bounds__(256) k_cvt_t(const float* __restrict__ w,
                                               unsigned short* __restrict__ wt) {
    __shared__ float tile[32][33];
    const int k0 = blockIdx.y * 32;
    const int n0 = blockIdx.x * 32;
    const int tx = threadIdx.x & 31;
    const int ty = threadIdx.x >> 5;
#pragma unroll
    for (int i = 0; i < 4; ++i)
        tile[ty + i * 8][tx] = w[(size_t)(k0 + ty + i * 8) * DM + n0 + tx];
    __syncthreads();
#pragma unroll
    for (int i = 0; i < 4; ++i)
        wt[(size_t)(n0 + ty + i * 8) * DM + k0 + tx] = f2bf(tile[tx][ty + i * 8]);
}

// ---------------------------------------------------------------------------
// QKV projection GEMM: C[4096x1024] = X_bf16 @ W (WT is N-major bf16)
//   blockIdx.y = 0/1/2 selects Q/K/V.  8 waves/block, 32x64 tile per wave:
//   per k-step 12 x 16B loads feed 8 independent WMMAs (loads batched first).
//   Q,K written [B,H,S,64]; V written transposed [B,H,64,S] via LDS transpose.
// ---------------------------------------------------------------------------
__global__ void __launch_bounds__(256) k_proj(
        const unsigned short* __restrict__ Xq, const unsigned short* __restrict__ Xk,
        const unsigned short* __restrict__ Xv,
        const unsigned short* __restrict__ WqT, const unsigned short* __restrict__ WkT,
        const unsigned short* __restrict__ WvT,
        unsigned short* __restrict__ Qb, unsigned short* __restrict__ Kb,
        unsigned short* __restrict__ Vt) {
    __shared__ unsigned short tlds[8][8 * 272];   // per-wave 16x16(+pad) transpose bufs

    const int sel = blockIdx.y;
    const unsigned short* X  = (sel == 0) ? Xq  : (sel == 1) ? Xk  : Xv;
    const unsigned short* WT = (sel == 0) ? WqT : (sel == 1) ? WkT : WvT;

    const int wave = threadIdx.x >> 5;
    const int lane = threadIdx.x & 31;
    const int ln = lane & 15, hl = lane >> 4;

    const int tile = blockIdx.x * 8 + wave;   // 2048 wave-tiles (128 M x 16 Ngroups)
    const int m0 = (tile >> 4) * 32;
    const int g  = tile & 15;                 // 64-wide N group == head index

    const v8f vzero = {0.f, 0.f, 0.f, 0.f, 0.f, 0.f, 0.f, 0.f};
    v8f acc[2][4];
#pragma unroll
    for (int mi = 0; mi < 2; ++mi)
#pragma unroll
        for (int t = 0; t < 4; ++t) acc[mi][t] = vzero;

    const unsigned short* arow0 = X + (size_t)(m0 + ln) * DM;
    const unsigned short* arow1 = X + (size_t)(m0 + 16 + ln) * DM;
    const unsigned short* brow  = WT + (size_t)(g * 64 + ln) * DM;

#pragma unroll 2
    for (int k0 = 0; k0 < DM; k0 += 32) {
        // batch ALL loads first -> distinct regs, overlapped with WMMA chain
        BF16Frag a0 = ldfragA(arow0 + k0 + hl * 8);
        BF16Frag a1 = ldfragA(arow1 + k0 + hl * 8);
        BF16Frag bf[4];
#pragma unroll
        for (int t = 0; t < 4; ++t)
            bf[t] = ldfragB(brow + (size_t)t * 16 * DM + k0 + hl * 8);
#pragma unroll
        for (int t = 0; t < 4; ++t) acc[0][t] = wmma_bf16(a0, bf[t], acc[0][t]);
#pragma unroll
        for (int t = 0; t < 4; ++t) acc[1][t] = wmma_bf16(a1, bf[t], acc[1][t]);
    }

    const int bb = m0 >> 11;                  // batch (32-row tile never crosses)
    const int s0 = m0 & 2047;

    if (sel == 2) {
        // transpose 16x16 subtiles through LDS so Vt stores are s-contiguous
        unsigned short* tl = &tlds[wave][0];
#pragma unroll
        for (int mi = 0; mi < 2; ++mi)
#pragma unroll
            for (int t = 0; t < 4; ++t)
#pragma unroll
                for (int j = 0; j < 8; ++j)
                    tl[(mi * 4 + t) * 272 + ln * 17 + j + 8 * hl] = f2bf(acc[mi][t][j]);
        __syncthreads();
#pragma unroll
        for (int mi = 0; mi < 2; ++mi)
#pragma unroll
            for (int t = 0; t < 4; ++t)
#pragma unroll
                for (int j = 0; j < 8; ++j) {
                    int dd = j + 8 * hl;
                    int d  = t * 16 + dd;
                    Vt[(((size_t)(bb * NH + g) * DHEAD) + d) * SEQ + s0 + mi * 16 + ln] =
                        tl[(mi * 4 + t) * 272 + dd * 17 + ln];
                }
    } else {
        unsigned short* dst = (sel == 0) ? Qb : Kb;
#pragma unroll
        for (int mi = 0; mi < 2; ++mi)
#pragma unroll
            for (int t = 0; t < 4; ++t)
#pragma unroll
                for (int j = 0; j < 8; ++j) {
                    int s = s0 + mi * 16 + j + 8 * hl;
                    int d = t * 16 + ln;
                    dst[(((size_t)(bb * NH + g) * SEQ) + s) * DHEAD + d] =
                        f2bf(acc[mi][t][j]);
                }
    }
}

// ---------------------------------------------------------------------------
// Fused attention with softmax over the HEADS axis.
//   16 waves (one head each), shared 16-query tile, k streamed in 32-steps.
//   V fragments loaded BEFORE the softmax so VMEM latency hides behind
//   exp + barriers + cross-head LDS reduction.
// ---------------------------------------------------------------------------
#define E_STRIDE  33
#define E_HEAD    (16 * E_STRIDE)      // 528
#define A_STRIDE  40
#define A_HEAD    (16 * A_STRIDE)      // 640

__global__ void __launch_bounds__(512) k_attn(
        const unsigned short* __restrict__ Qb, const unsigned short* __restrict__ Kb,
        const unsigned short* __restrict__ Vt, unsigned short* __restrict__ ctx) {
    __shared__ float          lds_e[NH * E_HEAD];        // exp(scores) per head
    __shared__ float          lds_r[16 * 32];            // 1 / sum_h exp
    __shared__ unsigned short lds_attn[NH * A_HEAD];     // normalized attn (bf16)

    const int b  = blockIdx.x >> 7;
    const int q0 = (blockIdx.x & 127) * 16;
    const int h    = threadIdx.x >> 5;
    const int lane = threadIdx.x & 31;
    const int ln = lane & 15, hl = lane >> 4;

    const unsigned short* Qh = Qb + (size_t)(b * NH + h) * SEQ * DHEAD;
    const unsigned short* Kh = Kb + (size_t)(b * NH + h) * SEQ * DHEAD;
    const unsigned short* Vh = Vt + (size_t)(b * NH + h) * DHEAD * SEQ;

    // Q tile A-fragments (fixed for whole k-loop): two d-halves of 32
    BF16Frag aq[2];
#pragma unroll
    for (int dh = 0; dh < 2; ++dh)
        aq[dh] = ldfragA(Qh + (size_t)(q0 + ln) * DHEAD + dh * 32 + hl * 8);

    const v8f vzero = {0.f, 0.f, 0.f, 0.f, 0.f, 0.f, 0.f, 0.f};
    v8f cacc[4] = {vzero, vzero, vzero, vzero};   // 16q x 64v context accumulators

    for (int k0 = 0; k0 < SEQ; k0 += 32) {
        if (k0 + 32 < SEQ) {   // CDNA5 global_prefetch_b8 for the next K/V tiles
            __builtin_prefetch(Kh + (size_t)(k0 + 32 + ln) * DHEAD, 0, 0);
            __builtin_prefetch(Vh + (size_t)ln * SEQ + k0 + 32, 0, 0);
        }

        // ---- scores: 16q x 32k, reduced over d=64; batch all 4 K B-frags first
        BF16Frag kb[4];
        kb[0] = ldfragB(Kh + (size_t)(k0 + ln) * DHEAD + hl * 8);           // dh0,k0-15
        kb[1] = ldfragB(Kh + (size_t)(k0 + 16 + ln) * DHEAD + hl * 8);      // dh0,k16-31
        kb[2] = ldfragB(Kh + (size_t)(k0 + ln) * DHEAD + 32 + hl * 8);      // dh1,k0-15
        kb[3] = ldfragB(Kh + (size_t)(k0 + 16 + ln) * DHEAD + 32 + hl * 8); // dh1,k16-31
        v8f s0 = wmma_bf16(aq[0], kb[0], vzero);
        v8f s1 = wmma_bf16(aq[0], kb[1], vzero);
        s0 = wmma_bf16(aq[1], kb[2], s0);
        s1 = wmma_bf16(aq[1], kb[3], s1);

        // ---- V fragments for this k-step: independent of softmax, load early
        BF16Frag bv[4];
#pragma unroll
        for (int vt = 0; vt < 4; ++vt)
            bv[vt] = ldfragB(Vh + (size_t)(vt * 16 + ln) * SEQ + k0 + hl * 8);

        // ---- exp(scores/8), stash in LDS for the cross-head reduction
        float e0[8], e1[8];
        float* le = lds_e + h * E_HEAD;
#pragma unroll
        for (int j = 0; j < 8; ++j) {
            int q = j + hl * 8;
            e0[j] = __expf(s0[j] * 0.125f);
            e1[j] = __expf(s1[j] * 0.125f);
            le[q * E_STRIDE + ln]      = e0[j];
            le[q * E_STRIDE + 16 + ln] = e1[j];
        }
        __syncthreads();

        // ---- softmax denominator over the 16 heads, per (q,k)
        {
            int t = threadIdx.x;              // t = q*32 + k  (512 positions)
            int q = t >> 5, k = t & 31;
            float sum = 0.f;
#pragma unroll
            for (int hh = 0; hh < NH; ++hh) sum += lds_e[hh * E_HEAD + q * E_STRIDE + k];
            lds_r[t] = __frcp_rn(sum);
        }
        __syncthreads();

        // ---- normalize, convert to bf16, lay out [q][k] for A-fragment reload
        unsigned short* la = lds_attn + h * A_HEAD;
#pragma unroll
        for (int j = 0; j < 8; ++j) {
            int q = j + hl * 8;
            la[q * A_STRIDE + ln]      = f2bf(e0[j] * lds_r[q * 32 + ln]);
            la[q * A_STRIDE + 16 + ln] = f2bf(e1[j] * lds_r[q * 32 + 16 + ln]);
        }
        __syncthreads();

        // ---- context += attn(16x32) @ V(32x64); V already in registers
        BF16Frag fa;
        const unsigned short* ar = la + ln * A_STRIDE;
        fa.u4[0] = *reinterpret_cast<const uint4*>(ar + hl * 8);
        fa.u4[1] = *reinterpret_cast<const uint4*>(ar + 16 + hl * 8);
#pragma unroll
        for (int vt = 0; vt < 4; ++vt)
            cacc[vt] = wmma_bf16(fa, bv[vt], cacc[vt]);
    }

    // ---- write context as bf16 [B*S, H*64] (natural layout for out-proj GEMM)
#pragma unroll
    for (int vt = 0; vt < 4; ++vt)
#pragma unroll
        for (int j = 0; j < 8; ++j) {
            int r = b * SEQ + q0 + j + hl * 8;
            int c = h * DHEAD + vt * 16 + ln;
            ctx[(size_t)r * DM + c] = f2bf(cacc[vt][j]);
        }
}

// ---------------------------------------------------------------------------
// Output projection + residual: outp = ctx @ Wo + input_Q   (f32 result)
//   32x64 tile per wave, same batched-load scheme as k_proj.
// ---------------------------------------------------------------------------
__global__ void __launch_bounds__(256) k_outproj(
        const unsigned short* __restrict__ ctx, const unsigned short* __restrict__ WoT,
        const float* __restrict__ resid, float* __restrict__ outp) {
    const int wave = threadIdx.x >> 5;
    const int lane = threadIdx.x & 31;
    const int ln = lane & 15, hl = lane >> 4;

    const int tile = blockIdx.x * 8 + wave;   // 2048 wave-tiles
    const int m0 = (tile >> 4) * 32;
    const int g  = tile & 15;

    const v8f vzero = {0.f, 0.f, 0.f, 0.f, 0.f, 0.f, 0.f, 0.f};
    v8f acc[2][4];
#pragma unroll
    for (int mi = 0; mi < 2; ++mi)
#pragma unroll
        for (int t = 0; t < 4; ++t) acc[mi][t] = vzero;

    const unsigned short* arow0 = ctx + (size_t)(m0 + ln) * DM;
    const unsigned short* arow1 = ctx + (size_t)(m0 + 16 + ln) * DM;
    const unsigned short* brow  = WoT + (size_t)(g * 64 + ln) * DM;

#pragma unroll 2
    for (int k0 = 0; k0 < DM; k0 += 32) {
        BF16Frag a0 = ldfragA(arow0 + k0 + hl * 8);
        BF16Frag a1 = ldfragA(arow1 + k0 + hl * 8);
        BF16Frag bf[4];
#pragma unroll
        for (int t = 0; t < 4; ++t)
            bf[t] = ldfragB(brow + (size_t)t * 16 * DM + k0 + hl * 8);
#pragma unroll
        for (int t = 0; t < 4; ++t) acc[0][t] = wmma_bf16(a0, bf[t], acc[0][t]);
#pragma unroll
        for (int t = 0; t < 4; ++t) acc[1][t] = wmma_bf16(a1, bf[t], acc[1][t]);
    }

#pragma unroll
    for (int mi = 0; mi < 2; ++mi)
#pragma unroll
        for (int t = 0; t < 4; ++t)
#pragma unroll
            for (int j = 0; j < 8; ++j) {
                int r = m0 + mi * 16 + j + 8 * hl;
                int c = g * 64 + t * 16 + ln;
                outp[(size_t)r * DM + c] = acc[mi][t][j] + resid[(size_t)r * DM + c];
            }
}

// ---------------------------------------------------------------------------
// Row-wise LayerNorm over 1024, one block per row
// ---------------------------------------------------------------------------
__global__ void __launch_bounds__(256) k_ln(
        const float* __restrict__ x, const float* __restrict__ gamma,
        const float* __restrict__ beta, float* __restrict__ out) {
    __shared__ float red[256];
    const int row = blockIdx.x;
    const float* xr = x + (size_t)row * DM;

    float v[4];
    float sum = 0.f;
#pragma unroll
    for (int i = 0; i < 4; ++i) { v[i] = xr[threadIdx.x + i * 256]; sum += v[i]; }
    red[threadIdx.x] = sum;
    __syncthreads();
    for (int s = 128; s > 0; s >>= 1) {
        if (threadIdx.x < s) red[threadIdx.x] += red[threadIdx.x + s];
        __syncthreads();
    }
    float mu = red[0] * (1.0f / DM);
    __syncthreads();

    float vs = 0.f;
#pragma unroll
    for (int i = 0; i < 4; ++i) { float d = v[i] - mu; vs += d * d; }
    red[threadIdx.x] = vs;
    __syncthreads();
    for (int s = 128; s > 0; s >>= 1) {
        if (threadIdx.x < s) red[threadIdx.x] += red[threadIdx.x + s];
        __syncthreads();
    }
    float rstd = rsqrtf(red[0] * (1.0f / DM) + 1e-5f);

#pragma unroll
    for (int i = 0; i < 4; ++i) {
        int c = threadIdx.x + i * 256;
        out[(size_t)row * DM + c] = (v[i] - mu) * rstd * gamma[c] + beta[c];
    }
}

// ---------------------------------------------------------------------------
extern "C" void kernel_launch(void* const* d_in, const int* in_sizes, int n_in,
                              void* d_out, int out_size, void* d_ws, size_t ws_size,
                              hipStream_t stream) {
    (void)in_sizes; (void)n_in; (void)out_size; (void)ws_size;
    const float* inQ   = (const float*)d_in[0];
    const float* inK   = (const float*)d_in[1];
    const float* inV   = (const float*)d_in[2];
    const float* Wq    = (const float*)d_in[3];
    const float* Wk    = (const float*)d_in[4];
    const float* Wv    = (const float*)d_in[5];
    const float* Wo    = (const float*)d_in[6];
    const float* gamma = (const float*)d_in[7];
    const float* beta  = (const float*)d_in[8];
    float* out = (float*)d_out;

    char* ws = (char*)d_ws;
    size_t off = 0;
    auto alloc = [&](size_t bytes) -> char* {
        char* p = ws + off;
        off += (bytes + 255) & ~(size_t)255;
        return p;
    };

    const size_t MROWS = (size_t)NB * SEQ;                       // 4096
    unsigned short* WqT = (unsigned short*)alloc((size_t)DM * DM * 2);
    unsigned short* WkT = (unsigned short*)alloc((size_t)DM * DM * 2);
    unsigned short* WvT = (unsigned short*)alloc((size_t)DM * DM * 2);
    unsigned short* WoT = (unsigned short*)alloc((size_t)DM * DM * 2);
    unsigned short* Xq  = (unsigned short*)alloc(MROWS * DM * 2);
    unsigned short* Xk  = (unsigned short*)alloc(MROWS * DM * 2);
    unsigned short* Xv  = (unsigned short*)alloc(MROWS * DM * 2);
    unsigned short* Qb  = (unsigned short*)alloc(MROWS * DM * 2);
    unsigned short* Kb  = (unsigned short*)alloc(MROWS * DM * 2);
    unsigned short* Vt  = (unsigned short*)alloc(MROWS * DM * 2);
    unsigned short* ctx = (unsigned short*)alloc(MROWS * DM * 2);
    float*          pre = (float*)alloc(MROWS * DM * 4);

    const int nAct = (int)(MROWS * DM);                          // 4 Mi
    k_cvt<<<dim3(nAct / (256 * 8)), 256, 0, stream>>>(inQ, Xq, nAct);
    k_cvt<<<dim3(nAct / (256 * 8)), 256, 0, stream>>>(inK, Xk, nAct);
    k_cvt<<<dim3(nAct / (256 * 8)), 256, 0, stream>>>(inV, Xv, nAct);

    k_cvt_t<<<dim3(32, 32), 256, 0, stream>>>(Wq, WqT);
    k_cvt_t<<<dim3(32, 32), 256, 0, stream>>>(Wk, WkT);
    k_cvt_t<<<dim3(32, 32), 256, 0, stream>>>(Wv, WvT);
    k_cvt_t<<<dim3(32, 32), 256, 0, stream>>>(Wo, WoT);

    // 2048 wave-tiles / 8 waves per block = 256 blocks; y = Q/K/V
    k_proj<<<dim3(256, 3), 256, 0, stream>>>(Xq, Xk, Xv, WqT, WkT, WvT, Qb, Kb, Vt);

    // one block per (batch, 16-query tile): 2 * 128 = 256 blocks, 16 waves each
    k_attn<<<dim3(NB * (SEQ / 16)), 512, 0, stream>>>(Qb, Kb, Vt, ctx);

    k_outproj<<<dim3(256), 256, 0, stream>>>(ctx, WoT, inQ, pre);

    k_ln<<<dim3((int)MROWS), 256, 0, stream>>>(pre, gamma, beta, out);
}
